// MulHeadCrossAttn_738734375600
// MI455X (gfx1250) — matrix operand order repaired and checked by reading, not compiled, not run on hardware
//
#include <hip/hip_runtime.h>

typedef __bf16 bf16_t;
typedef __attribute__((ext_vector_type(16))) __bf16 v16bf;
typedef __attribute__((ext_vector_type(8)))  float  v8f;
typedef __attribute__((__vector_size__(16))) int i32x4;

#define LN10000 9.210340371976184f

// ---------------------------------------------------------------------------
// CDNA5 async global->LDS copy (ASYNCcnt-tracked), with sync fallback
// ---------------------------------------------------------------------------
#if defined(__AMDGCN__) && __has_builtin(__builtin_amdgcn_global_load_async_to_lds_b128)
#define ASYNC_G2L 1
#else
#define ASYNC_G2L 0
#endif

__device__ __forceinline__ void copy16_g2l(const bf16_t* __restrict__ g,
                                           bf16_t* l) {
#if ASYNC_G2L
  __builtin_amdgcn_global_load_async_to_lds_b128(
      (__attribute__((address_space(1))) i32x4*)g,
      (__attribute__((address_space(3))) i32x4*)l, 0, 0);
#else
  *(uint4*)l = *(const uint4*)g;
#endif
}

__device__ __forceinline__ void wait_async_all() {
#if ASYNC_G2L
#if __has_builtin(__builtin_amdgcn_s_wait_asynccnt)
  __builtin_amdgcn_s_wait_asynccnt(0);
#else
  asm volatile("s_wait_asynccnt 0x0" ::: "memory");
#endif
#endif
}

// ---------------------------------------------------------------------------
// DPP16 cross-lane reductions within a 16-lane row (wave32: rows = lane/16)
// ---------------------------------------------------------------------------
template <int CTRL>
__device__ __forceinline__ float dpp_movf(float x) {
  return __int_as_float(__builtin_amdgcn_update_dpp(
      0, __float_as_int(x), CTRL, 0xF, 0xF, true));
}

__device__ __forceinline__ float row16_max(float x) {
  x = fmaxf(x, dpp_movf<0xB1>(x));    // quad_perm [1,0,3,2]  (xor 1)
  x = fmaxf(x, dpp_movf<0x4E>(x));    // quad_perm [2,3,0,1]  (xor 2)
  x = fmaxf(x, dpp_movf<0x141>(x));   // row_half_mirror
  x = fmaxf(x, dpp_movf<0x140>(x));   // row_mirror
  return x;
}

__device__ __forceinline__ float row16_sum(float x) {
  x += dpp_movf<0xB1>(x);
  x += dpp_movf<0x4E>(x);
  x += dpp_movf<0x141>(x);
  x += dpp_movf<0x140>(x);
  return x;
}

// ---------------------------------------------------------------------------
// Fragment layout helpers (CDNA5 ISA 7.12.2, wave32)
// ---------------------------------------------------------------------------
__device__ __forceinline__ int a_koff(int j, int hi) {
  // 16-bit A 16x32 as used here (B matches): lanes 0-15: K=0..15, lanes 16-31: K=16..31 (contiguous)
  return 16 * hi + 2 * j;
}

__device__ __forceinline__ v8f wmma_bf16(v16bf a, v16bf b, v8f c) {
  return __builtin_amdgcn_wmma_f32_16x16x32_bf16(false, a, false, b, (short)0, c,
                                                 false, false);
}

__device__ __forceinline__ float rope_val(float x, float other, int s, int d) {
  float inv = __expf(-(float)(d & 63) * (1.0f / 64.0f) * LN10000);
  float ang = (float)s * inv;
  float sn, cs;
  __sincosf(ang, &sn, &cs);
  return (d < 64) ? (x * cs - other * sn) : (x * cs + other * sn);
}

// ---------------------------------------------------------------------------
// f32 -> bf16 conversion (grid-stride)
// ---------------------------------------------------------------------------
__global__ void f32_to_bf16_kernel(const float* __restrict__ in,
                                   bf16_t* __restrict__ out, int n) {
  int i = blockIdx.x * blockDim.x + threadIdx.x;
  int stride = gridDim.x * blockDim.x;
  for (; i < n; i += stride) out[i] = (bf16_t)in[i];
}

// ---------------------------------------------------------------------------
// GEMM: C[M,N] = A[M,K](bf16) * B[K,N](bf16), double-buffered async LDS
// 128x128 block tile, 8 waves, each wave 32x64 via 2x4 WMMA 16x16x32 bf16
// ---------------------------------------------------------------------------
template <bool OUT_BF16>
__global__ __launch_bounds__(256) void gemm_wmma_bf16(
    const bf16_t* __restrict__ A, const bf16_t* __restrict__ B,
    void* __restrict__ C, int M, int N, int K) {
  __shared__ bf16_t As[2][128][40];   // 80B rows: 16B-aligned, 20-bank stride
  __shared__ bf16_t Bs[2][32][136];   // 272B rows: 16B-aligned

  const int tid = threadIdx.x;
  const int lane = tid & 31;
  const int wave = tid >> 5;
  const int hi = lane >> 4;
  const int lo = lane & 15;
  const int wm = wave & 3;
  const int wn = wave >> 2;
  const int bm = blockIdx.y * 128;
  const int bn = blockIdx.x * 128;

  v8f acc[2][4];
#pragma unroll
  for (int mi = 0; mi < 2; ++mi)
#pragma unroll
    for (int ni = 0; ni < 4; ++ni)
#pragma unroll
      for (int r = 0; r < 8; ++r) acc[mi][ni][r] = 0.0f;

  const int arow = tid >> 1, acol = (tid & 1) * 16;
  const int brow = tid >> 3, bcol = (tid & 7) * 16;

  auto issue_tile = [&](int k0, int buf) {
    const bf16_t* ag = A + (size_t)(bm + arow) * K + k0 + acol;
    copy16_g2l(ag, &As[buf][arow][acol]);
    copy16_g2l(ag + 8, &As[buf][arow][acol + 8]);
    const bf16_t* bg = B + (size_t)(k0 + brow) * N + bn + bcol;
    copy16_g2l(bg, &Bs[buf][brow][bcol]);
    copy16_g2l(bg + 8, &Bs[buf][brow][bcol + 8]);
  };

  issue_tile(0, 0);
  int cur = 0;
  for (int k0 = 0; k0 < K; k0 += 32) {
    wait_async_all();
    __syncthreads();
    if (k0 + 32 < K) issue_tile(k0 + 32, cur ^ 1);

    v16bf afrag[2];
#pragma unroll
    for (int mi = 0; mi < 2; ++mi) {
      const int row = 32 * wm + 16 * mi + lo;
#pragma unroll
      for (int j = 0; j < 8; ++j) {
        const int kk = a_koff(j, hi);
        afrag[mi][2 * j]     = As[cur][row][kk];
        afrag[mi][2 * j + 1] = As[cur][row][kk + 1];
      }
    }
    v16bf bfrag[4];
#pragma unroll
    for (int ni = 0; ni < 4; ++ni) {
      const int col = 64 * wn + 16 * ni + lo;
#pragma unroll
      for (int j = 0; j < 8; ++j) {
        const int kk = 16 * hi + 2 * j;
        bfrag[ni][2 * j]     = Bs[cur][kk][col];
        bfrag[ni][2 * j + 1] = Bs[cur][kk + 1][col];
      }
    }
#pragma unroll
    for (int mi = 0; mi < 2; ++mi)
#pragma unroll
      for (int ni = 0; ni < 4; ++ni)
        acc[mi][ni] = wmma_bf16(afrag[mi], bfrag[ni], acc[mi][ni]);
    cur ^= 1;
  }

#pragma unroll
  for (int mi = 0; mi < 2; ++mi)
#pragma unroll
    for (int ni = 0; ni < 4; ++ni) {
      const int col = bn + 64 * wn + 16 * ni + lo;
#pragma unroll
      for (int r = 0; r < 8; ++r) {
        const int row = bm + 32 * wm + 16 * mi + 8 * hi + r;
        if (OUT_BF16)
          ((bf16_t*)C)[(size_t)row * N + col] = (bf16_t)acc[mi][ni][r];
        else
          ((float*)C)[(size_t)row * N + col] = acc[mi][ni][r];
      }
    }
}

// ---------------------------------------------------------------------------
// RoPE on Q: (B,S,H*128) bf16 -> (B,H,S,128) bf16
// ---------------------------------------------------------------------------
__global__ void rope_q_kernel(const bf16_t* __restrict__ Qraw,
                              bf16_t* __restrict__ Qb) {
  const int idx = blockIdx.x * blockDim.x + threadIdx.x;
  const int d = idx & 127;
  const int h = (idx >> 7) & 15;
  const int s = (idx >> 11) & 2047;
  const int b = idx >> 22;
  const bf16_t* row = Qraw + (size_t)(b * 2048 + s) * 2048 + h * 128;
  const float x = (float)row[d];
  const float other = (float)row[(d < 64) ? d + 64 : d - 64];
  Qb[(((size_t)(b * 16 + h)) * 2048 + s) * 128 + d] =
      (bf16_t)rope_val(x, other, s, d);
}

// ---------------------------------------------------------------------------
// Split KV (B,S,4096) -> K (B,H,S,128) (copy) and V (B,H,S,128) (RoPE)
// ---------------------------------------------------------------------------
__global__ void split_rope_kv_kernel(const bf16_t* __restrict__ KVraw,
                                     bf16_t* __restrict__ Kb,
                                     bf16_t* __restrict__ Vb) {
  const int idx = blockIdx.x * blockDim.x + threadIdx.x;
  const int c = idx & 2047;
  const int s = (idx >> 11) & 2047;
  const int b = idx >> 22;
  const int h = c >> 7;
  const int d = c & 127;
  const bf16_t* row = KVraw + (size_t)(b * 2048 + s) * 4096;
  const size_t o = (((size_t)(b * 16 + h)) * 2048 + s) * 128 + d;
  Kb[o] = row[c];
  const float x = (float)row[2048 + c];
  const float other = (float)row[2048 + h * 128 + ((d < 64) ? d + 64 : d - 64)];
  Vb[o] = (bf16_t)rope_val(x, other, s, d);
}

// ---------------------------------------------------------------------------
// Flash attention per (b, h): 128 q-rows per block (16 per wave), 32-key
// blocks double-buffered through LDS, DPP-based online softmax, WMMA for
// QK^T and P*V. One workgroup barrier per key block.
// ---------------------------------------------------------------------------
__global__ __launch_bounds__(256) void attn_kernel(
    const bf16_t* __restrict__ Qb, const bf16_t* __restrict__ Kb,
    const bf16_t* __restrict__ Vb, bf16_t* __restrict__ Ob) {
  __shared__ bf16_t Ks[2][32][136];
  __shared__ bf16_t Vs[2][32][136];
  __shared__ bf16_t Ps[8][16][32];

  const int tid = threadIdx.x;
  const int lane = tid & 31;
  const int wave = tid >> 5;
  const int hi = lane >> 4;
  const int lo = lane & 15;
  const int b = blockIdx.z, h = blockIdx.y;
  const int q0 = blockIdx.x * 128;

  const bf16_t* Qh = Qb + ((size_t)(b * 16 + h)) * 2048 * 128;
  const bf16_t* Kh = Kb + ((size_t)(b * 16 + h)) * 2048 * 128;
  const bf16_t* Vh = Vb + ((size_t)(b * 16 + h)) * 2048 * 128;

  // Q fragments for this wave's 16 rows (registers for the whole pass)
  const int qrow = q0 + wave * 16 + lo;
  v16bf qf[4];
#pragma unroll
  for (int c = 0; c < 4; ++c)
#pragma unroll
    for (int j = 0; j < 8; ++j) {
      const int kk = 32 * c + a_koff(j, hi);
      qf[c][2 * j]     = Qh[(size_t)qrow * 128 + kk];
      qf[c][2 * j + 1] = Qh[(size_t)qrow * 128 + kk + 1];
    }

  v8f o[8];
#pragma unroll
  for (int t = 0; t < 8; ++t)
#pragma unroll
    for (int r = 0; r < 8; ++r) o[t][r] = 0.0f;

  float mrow[8], lrow[8];
#pragma unroll
  for (int r = 0; r < 8; ++r) { mrow[r] = -1e30f; lrow[r] = 0.0f; }

  const float scale = 0.08838834764831845f;  // 1/sqrt(128)
  const int krow = tid >> 3, kcol = (tid & 7) * 16;

  auto issue_kv = [&](int kbase, int buf) {
    const bf16_t* kg = Kh + (size_t)(kbase + krow) * 128 + kcol;
    copy16_g2l(kg, &Ks[buf][krow][kcol]);
    copy16_g2l(kg + 8, &Ks[buf][krow][kcol + 8]);
    const bf16_t* vg = Vh + (size_t)(kbase + krow) * 128 + kcol;
    copy16_g2l(vg, &Vs[buf][krow][kcol]);
    copy16_g2l(vg + 8, &Vs[buf][krow][kcol + 8]);
  };

  issue_kv(0, 0);
  int cur = 0;
  for (int k0 = 0; k0 < 2048; k0 += 32) {
    wait_async_all();
    __syncthreads();
    if (k0 + 32 < 2048) issue_kv(k0 + 32, cur ^ 1);

    // scores S = Q * K^T : two 16x16 tiles (keys +0..15, +16..31)
    v8f s[2];
#pragma unroll
    for (int t = 0; t < 2; ++t) {
#pragma unroll
      for (int r = 0; r < 8; ++r) s[t][r] = 0.0f;
#pragma unroll
      for (int c = 0; c < 4; ++c) {
        v16bf bf;
#pragma unroll
        for (int j = 0; j < 8; ++j) {
          const int kd = 32 * c + 16 * hi + 2 * j;
          bf[2 * j]     = Ks[cur][16 * t + lo][kd];
          bf[2 * j + 1] = Ks[cur][16 * t + lo][kd + 1];
        }
        s[t] = wmma_bf16(qf[c], bf, s[t]);
      }
    }

    // online softmax: each C-row lives across one 16-lane DPP row
#pragma unroll
    for (int r = 0; r < 8; ++r) {
      float m_ = row16_max(fmaxf(s[0][r], s[1][r]) * scale);
      const float mnew = fmaxf(mrow[r], m_);
      const float alpha = __expf(mrow[r] - mnew);
      mrow[r] = mnew;
      const float p0 = __expf(s[0][r] * scale - mnew);
      const float p1 = __expf(s[1][r] * scale - mnew);
      lrow[r] = lrow[r] * alpha + row16_sum(p0 + p1);
      Ps[wave][8 * hi + r][lo]      = (bf16_t)p0;
      Ps[wave][8 * hi + r][16 + lo] = (bf16_t)p1;
#pragma unroll
      for (int t = 0; t < 8; ++t) o[t][r] *= alpha;
    }
    // Ps is wave-private; DS ops from one wave are processed in order, so a
    // dscnt drain + compiler fence suffices (no workgroup barrier needed).
    asm volatile("s_wait_dscnt 0x0" ::: "memory");

    // P (16x32) as A-fragment, V block as B-fragments; O += P*V
    v16bf pf;
#pragma unroll
    for (int j = 0; j < 8; ++j) {
      const int kk = a_koff(j, hi);
      pf[2 * j]     = Ps[wave][lo][kk];
      pf[2 * j + 1] = Ps[wave][lo][kk + 1];
    }
#pragma unroll
    for (int t = 0; t < 8; ++t) {
      v16bf vf;
#pragma unroll
      for (int j = 0; j < 8; ++j) {
        const int kd = 16 * hi + 2 * j;
        vf[2 * j]     = Vs[cur][kd][16 * t + lo];
        vf[2 * j + 1] = Vs[cur][kd + 1][16 * t + lo];
      }
      o[t] = wmma_bf16(pf, vf, o[t]);
    }
    cur ^= 1;
  }

  // normalize and store to (B, S, H*128) layout for the output projection
#pragma unroll
  for (int r = 0; r < 8; ++r) {
    const float inv = 1.0f / lrow[r];
    const int row = q0 + wave * 16 + 8 * hi + r;
#pragma unroll
    for (int t = 0; t < 8; ++t) {
      Ob[((size_t)(b * 2048) + row) * 2048 + h * 128 + 16 * t + lo] =
          (bf16_t)(o[t][r] * inv);
    }
  }
}

// ---------------------------------------------------------------------------
// Launch orchestration
// ---------------------------------------------------------------------------
extern "C" void kernel_launch(void* const* d_in, const int* in_sizes, int n_in,
                              void* d_out, int out_size, void* d_ws,
                              size_t ws_size, hipStream_t stream) {
  (void)in_sizes; (void)n_in; (void)out_size; (void)ws_size;
  constexpr int B = 2, C1 = 2048, C2 = 2048;
  constexpr int DQ = 2048, DKV = 1024, DM = 2048, H = 16;

  const float* x1  = (const float*)d_in[0];
  const float* x2  = (const float*)d_in[1];
  const float* Wq  = (const float*)d_in[2];
  const float* Wkv = (const float*)d_in[3];
  const float* Wo  = (const float*)d_in[4];
  float* out = (float*)d_out;

  char* ws = (char*)d_ws;
  size_t off = 0;
  auto alloc = [&](size_t elems) -> bf16_t* {
    bf16_t* p = (bf16_t*)(ws + off);
    off += (elems * sizeof(bf16_t) + 255) & ~(size_t)255;
    return p;
  };

  bf16_t* bx1   = alloc((size_t)B * C1 * DQ);
  bf16_t* bx2   = alloc((size_t)B * C2 * DKV);
  bf16_t* bWq   = alloc((size_t)DQ * DM);
  bf16_t* bWkv  = alloc((size_t)DKV * 2 * DM);
  bf16_t* bWo   = alloc((size_t)DM * DQ);
  bf16_t* Qraw  = alloc((size_t)B * C1 * DM);
  bf16_t* KVraw = alloc((size_t)B * C2 * 2 * DM);
  bf16_t* Qb    = alloc((size_t)B * H * C1 * 128);
  bf16_t* Kb    = alloc((size_t)B * H * C2 * 128);
  bf16_t* Vb    = alloc((size_t)B * H * C2 * 128);
  bf16_t* Ob    = alloc((size_t)B * C1 * DM);

  auto cvt = [&](const float* in, bf16_t* outp, int n) {
    int blocks = (n + 255) / 256;
    if (blocks > 8192) blocks = 8192;
    f32_to_bf16_kernel<<<dim3(blocks), dim3(256), 0, stream>>>(in, outp, n);
  };
  cvt(x1, bx1, B * C1 * DQ);
  cvt(x2, bx2, B * C2 * DKV);
  cvt(Wq, bWq, DQ * DM);
  cvt(Wkv, bWkv, DKV * 2 * DM);
  cvt(Wo, bWo, DM * DQ);

  // Q = x1 @ Wq : [4096,2048]
  gemm_wmma_bf16<true><<<dim3(DM / 128, (B * C1) / 128), dim3(256), 0, stream>>>(
      bx1, bWq, Qraw, B * C1, DM, DQ);
  // KV = x2 @ Wkv : [4096,4096]
  gemm_wmma_bf16<true><<<dim3((2 * DM) / 128, (B * C2) / 128), dim3(256), 0,
                         stream>>>(bx2, bWkv, KVraw, B * C2, 2 * DM, DKV);

  rope_q_kernel<<<dim3((B * C1 * DM) / 256), dim3(256), 0, stream>>>(Qraw, Qb);
  split_rope_kv_kernel<<<dim3((B * C2 * DM) / 256), dim3(256), 0, stream>>>(
      KVraw, Kb, Vb);

  attn_kernel<<<dim3(C1 / 128, H, B), dim3(256), 0, stream>>>(Qb, Kb, Vb, Ob);

  // out = O @ Wo : [4096,2048] f32
  gemm_wmma_bf16<false><<<dim3(DQ / 128, (B * C1) / 128), dim3(256), 0, stream>>>(
      Ob, bWo, out, B * C1, DQ, DM);
}
